// Attention_43190191129190
// MI455X (gfx1250) — matrix operand order repaired and checked by reading, not compiled, not run on hardware
//
#include <hip/hip_runtime.h>

typedef __attribute__((ext_vector_type(16))) _Float16 v16h;
typedef __attribute__((ext_vector_type(8)))  float    v8f;

#define WMMA_F16(a, b, c) \
  __builtin_amdgcn_wmma_f32_16x16x32_f16(false, (a), false, (b), (short)0, (c), false, false)

// ---------------------------------------------------------------------------
// CDNA5 async copy: global -> LDS, 16 bytes per lane, tracked by ASYNCcnt.
// (CDNA5 ISA op 98 / CDNA5 ISA)
// ---------------------------------------------------------------------------
__device__ __forceinline__ void async_copy_b128(uint32_t lds_byte_off,
                                                const _Float16* gaddr) {
  asm volatile("global_load_async_to_lds_b128 %0, %1, off"
               :: "v"(lds_byte_off), "v"(gaddr)
               : "memory");
}

// ---------------------------------------------------------------------------
// Fragment loaders (wave32 WMMA operands in contiguous k order k = 16*(lane/16) + s, A and B alike; CDNA5 ISA)
// ---------------------------------------------------------------------------

// A-matrix 16x32 f16 from row-major X[ld] at tile (m0, k0).
__device__ __forceinline__ v16h load_a_frag(const _Float16* __restrict__ X,
                                            int ld, int m0, int k0, int lane) {
  const int m  = m0 + (lane & 15);
  const int hi = lane >> 4;
  const _Float16* row = X + (size_t)m * ld + k0;
  v16h a;
#pragma unroll
  for (int t = 0; t < 16; ++t) a[t] = row[hi * 16 + t];
  return a;
}

// B-matrix 32x16 f16 where B[k][n] = M[n][k], M row-major (i.e. X*M^T).
__device__ __forceinline__ v16h load_b_fragT(const _Float16* __restrict__ M,
                                             int ld, int k0, int n0, int lane) {
  const int n  = n0 + (lane & 15);
  const int kb = k0 + ((lane >> 4) << 4);
  const _Float16* p = M + (size_t)n * ld + kb;
  v16h b;
#pragma unroll
  for (int t = 0; t < 16; ++t) b[t] = p[t];
  return b;
}

// ---------------------------------------------------------------------------
// 1) Depthwise 3x3 conv + residual positional embedding, f32 -> f16
// ---------------------------------------------------------------------------
__global__ __launch_bounds__(256) void poscnn_kernel(
    const float* __restrict__ x, const float* __restrict__ cw,
    const float* __restrict__ cb, _Float16* __restrict__ xp) {
  const int idx = blockIdx.x * blockDim.x + threadIdx.x;  // (b*1024+n)*256+c
  if (idx >= 8 * 1024 * 256) return;
  const int c = idx & 255;
  const int n = (idx >> 8) & 1023;
  const int b = idx >> 18;
  const int h = n >> 5, w = n & 31;
  float s = cb[c];
  const float* wr = cw + c * 9;
#pragma unroll
  for (int dh = -1; dh <= 1; ++dh) {
#pragma unroll
    for (int dw = -1; dw <= 1; ++dw) {
      const int hh = h + dh, ww = w + dw;
      if (hh >= 0 && hh < 32 && ww >= 0 && ww < 32) {
        const int nn = (hh << 5) + ww;
        s += x[(((size_t)b << 10) + nn) * 256 + c] * wr[(dh + 1) * 3 + (dw + 1)];
      }
    }
  }
  s += x[idx];  // residual
  xp[idx] = (_Float16)s;
}

// ---------------------------------------------------------------------------
// 2) Weight conversion f32 -> f16
// ---------------------------------------------------------------------------
__global__ __launch_bounds__(256) void cvt_kernel(
    const float* __restrict__ a, int na, const float* __restrict__ b, int nb,
    _Float16* __restrict__ oa, _Float16* __restrict__ ob) {
  const int i = blockIdx.x * blockDim.x + threadIdx.x;
  if (i < na) oa[i] = (_Float16)a[i];
  else if (i - na < nb) ob[i - na] = (_Float16)b[i - na];
}

// ---------------------------------------------------------------------------
// 3) QKV GEMM: [8192,256]f16 x [768,256]^T -> q,k row-major [bh][n][d],
//    v transposed [bh][d][n].  One wave = 16x64 tile, K loop 256/32.
// ---------------------------------------------------------------------------
__global__ __launch_bounds__(256) void qkv_gemm_kernel(
    const _Float16* __restrict__ X, const _Float16* __restrict__ W,
    _Float16* __restrict__ qb, _Float16* __restrict__ kb,
    _Float16* __restrict__ vT) {
  const int lane = threadIdx.x & 31;
  const int wave = blockIdx.x * (blockDim.x >> 5) + (threadIdx.x >> 5);
  if (wave >= 512 * 12) return;
  const int mt = wave / 12, ng = wave % 12;
  const int m0 = mt << 4, n0base = ng << 6;

  v8f acc[4] = {v8f{}, v8f{}, v8f{}, v8f{}};
  for (int k0 = 0; k0 < 256; k0 += 32) {
    const v16h a = load_a_frag(X, 256, m0, k0, lane);
#pragma unroll
    for (int t = 0; t < 4; ++t) {
      const v16h b = load_b_fragT(W, 256, k0, n0base + (t << 4), lane);
      acc[t] = WMMA_F16(a, b, acc[t]);
    }
  }
  const int hi = lane >> 4, nn = lane & 15;
#pragma unroll
  for (int t = 0; t < 4; ++t) {
    const int col0 = n0base + (t << 4);
#pragma unroll
    for (int j = 0; j < 8; ++j) {
      const int row = m0 + j + (hi << 3);   // b*1024 + n
      const int col = col0 + nn;            // 0..767
      const int which = col >> 8;           // 0=q 1=k 2=v
      const int c = col & 255, hd = c >> 5, d = c & 31;
      const int b_ = row >> 10, n_ = row & 1023;
      const int bh = (b_ << 3) + hd;
      const _Float16 val = (_Float16)acc[t][j];
      if (which == 0)      qb[(((size_t)bh << 10) + n_) * 32 + d] = val;
      else if (which == 1) kb[(((size_t)bh << 10) + n_) * 32 + d] = val;
      else                 vT[(((size_t)bh << 5) + d) * 1024 + n_] = val;
    }
  }
}

// ---------------------------------------------------------------------------
// 4) Fused flash attention per (b,h).  Block = 4 waves sharing one (b,h);
//    K/V^T tiles (2KB each) staged to LDS via async-to-LDS b128 copies,
//    double-buffered so DMA of tile i+1 overlaps WMMA+softmax of tile i.
// ---------------------------------------------------------------------------
__global__ __launch_bounds__(128) void attn_kernel(
    const _Float16* __restrict__ qb, const _Float16* __restrict__ kb,
    const _Float16* __restrict__ vT, _Float16* __restrict__ ob) {
  __shared__ _Float16 lds_k[2][32 * 32];   // [buf][key-row][d]
  __shared__ _Float16 lds_v[2][32 * 32];   // [buf][d][key]
  __shared__ _Float16 p_lds[4][16 * 32];   // per-wave P staging

  const int tid   = threadIdx.x;
  const int lane  = tid & 31;
  const int wslot = tid >> 5;
  const int bh    = blockIdx.x >> 4;                       // 0..63
  const int i0    = (((blockIdx.x & 15) << 2) + wslot) << 4;  // query row tile

  const _Float16* Q = qb + ((size_t)bh << 10) * 32;
  const _Float16* K = kb + ((size_t)bh << 10) * 32;
  const _Float16* V = vT + ((size_t)bh << 5) * 1024;

  // cooperative copy geometry: 128 threads x 16B = 2KB per matrix per tile
  const int r  = tid >> 2;          // 0..31: key-row (K) / d-row (V^T)
  const int ch = (tid & 3) << 3;    // half-offset within row: 0,8,16,24
  const uint32_t lkb[2] = {(uint32_t)(uintptr_t)&lds_k[0][0],
                           (uint32_t)(uintptr_t)&lds_k[1][0]};
  const uint32_t lvb[2] = {(uint32_t)(uintptr_t)&lds_v[0][0],
                           (uint32_t)(uintptr_t)&lds_v[1][0]};
  const uint32_t chunk = (uint32_t)((r * 32 + ch) * 2);    // byte offset

  // prologue: issue tile 0 DMA
  async_copy_b128(lkb[0] + chunk, K + (size_t)r * 32 + ch);
  async_copy_b128(lvb[0] + chunk, V + (size_t)r * 1024 + ch);

  const v16h aq = load_a_frag(Q, 32, i0, 0, lane);
  // fold scale * log2(e): softmax runs in exp2 domain (v_exp_f32 direct)
  const float scl = 0.17677669529663689f * 1.4426950408889634f;

  float mrun[8], lrun[8];
#pragma unroll
  for (int j = 0; j < 8; ++j) { mrun[j] = -1e30f; lrun[j] = 0.0f; }
  v8f acc0 = {}, acc1 = {};
  const int hi = lane >> 4, nn = lane & 15;
  _Float16* P = p_lds[wslot];

  for (int it = 0; it < 32; ++it) {
    const int cur = it & 1;
    // issue next tile into the other buffer, then wait for current tile
    if (it + 1 < 32) {
      const int j1 = (it + 1) << 5;
      async_copy_b128(lkb[cur ^ 1] + chunk, K + (size_t)(j1 + r) * 32 + ch);
      async_copy_b128(lvb[cur ^ 1] + chunk, V + (size_t)r * 1024 + j1 + ch);
      asm volatile("s_wait_asynccnt 0x2" ::: "memory");  // 2 newest may fly
    } else {
      asm volatile("s_wait_asynccnt 0x0" ::: "memory");
    }
    __syncthreads();  // tile `cur` visible to all 4 waves

    const _Float16* lk = (const _Float16*)&lds_k[cur][0];
    const _Float16* lv = (const _Float16*)&lds_v[cur][0];
    const v16h bk0 = load_b_fragT(lk, 32, 0, 0, lane);
    const v16h bk1 = load_b_fragT(lk, 32, 0, 16, lane);
    const v16h bv0 = load_b_fragT(lv, 32, 0, 0, lane);
    const v16h bv1 = load_b_fragT(lv, 32, 0, 16, lane);

    // ---- S = Q K^T ----
    v8f s0 = {}, s1 = {};
    s0 = WMMA_F16(aq, bk0, s0);
    s1 = WMMA_F16(aq, bk1, s1);

    // ---- streaming softmax (exp2 domain) ----
#pragma unroll
    for (int j = 0; j < 8; ++j) {
      const float v0 = s0[j] * scl, v1 = s1[j] * scl;
      float mx = fmaxf(v0, v1);
#pragma unroll
      for (int off = 1; off <= 8; off <<= 1)
        mx = fmaxf(mx, __shfl_xor(mx, off, 32));   // 16-lane half reduce
      const float mnew = fmaxf(mrun[j], mx);
      const float alpha = exp2f(mrun[j] - mnew);
      const float p0 = exp2f(v0 - mnew), p1 = exp2f(v1 - mnew);
      float rs = p0 + p1;
#pragma unroll
      for (int off = 1; off <= 8; off <<= 1)
        rs += __shfl_xor(rs, off, 32);
      lrun[j] = lrun[j] * alpha + rs;
      mrun[j] = mnew;
      acc0[j] *= alpha;
      acc1[j] *= alpha;
      s0[j] = p0;
      s1[j] = p1;
    }

    // ---- D-layout -> A-layout via wave-private LDS tile ----
#pragma unroll
    for (int j = 0; j < 8; ++j) {
      P[(j + (hi << 3)) * 32 + nn]      = (_Float16)s0[j];
      P[(j + (hi << 3)) * 32 + 16 + nn] = (_Float16)s1[j];
    }
    // wave-private: per-wave DS ops are in order; fence compiler + DScnt
    asm volatile("s_wait_dscnt 0x0" ::: "memory");
    const v16h ap = load_a_frag(P, 32, 0, 0, lane);

    // ---- O += P V ----
    acc0 = WMMA_F16(ap, bv0, acc0);
    acc1 = WMMA_F16(ap, bv1, acc1);

    __syncthreads();  // all lk/lv reads done before buffer reuse
  }

  // ---- epilogue: normalize, store [B, N, H*d] f16 ----
  const int h = bh & 7, b_ = bh >> 3;
#pragma unroll
  for (int j = 0; j < 8; ++j) {
    const int n_ = i0 + j + (hi << 3);
    const size_t row = ((size_t)b_ << 10) + n_;
    const float inv = 1.0f / lrun[j];
    ob[row * 256 + h * 32 + nn]      = (_Float16)(acc0[j] * inv);
    ob[row * 256 + h * 32 + 16 + nn] = (_Float16)(acc1[j] * inv);
  }
}

// ---------------------------------------------------------------------------
// 5) Output projection: [8192,256]f16 x [256,256]^T + bias -> f32 d_out
// ---------------------------------------------------------------------------
__global__ __launch_bounds__(256) void out_gemm_kernel(
    const _Float16* __restrict__ O, const _Float16* __restrict__ W,
    const float* __restrict__ bias, float* __restrict__ out) {
  const int lane = threadIdx.x & 31;
  const int wave = blockIdx.x * (blockDim.x >> 5) + (threadIdx.x >> 5);
  if (wave >= 512 * 4) return;
  const int mt = wave >> 2, ng = wave & 3;
  const int m0 = mt << 4, n0base = ng << 6;

  v8f acc[4] = {v8f{}, v8f{}, v8f{}, v8f{}};
  for (int k0 = 0; k0 < 256; k0 += 32) {
    const v16h a = load_a_frag(O, 256, m0, k0, lane);
#pragma unroll
    for (int t = 0; t < 4; ++t) {
      const v16h b = load_b_fragT(W, 256, k0, n0base + (t << 4), lane);
      acc[t] = WMMA_F16(a, b, acc[t]);
    }
  }
  const int hi = lane >> 4, nn = lane & 15;
#pragma unroll
  for (int t = 0; t < 4; ++t) {
    const int col = n0base + (t << 4) + nn;
#pragma unroll
    for (int j = 0; j < 8; ++j) {
      const int row = m0 + j + (hi << 3);
      out[(size_t)row * 256 + col] = acc[t][j] + bias[col];
    }
  }
}

// ---------------------------------------------------------------------------
extern "C" void kernel_launch(void* const* d_in, const int* in_sizes, int n_in,
                              void* d_out, int out_size, void* d_ws,
                              size_t ws_size, hipStream_t stream) {
  const float* x      = (const float*)d_in[0];  // [8,1024,256]
  const float* conv_w = (const float*)d_in[1];  // [256,1,3,3]
  const float* conv_b = (const float*)d_in[2];  // [256]
  const float* qkv_w  = (const float*)d_in[3];  // [768,256]
  const float* out_w  = (const float*)d_in[4];  // [256,256]
  const float* out_b  = (const float*)d_in[5];  // [256]
  float* out = (float*)d_out;                   // [8,1024,256]

  char* ws = (char*)d_ws;
  _Float16* xp   = (_Float16*)(ws);                          // 8192*256   (4MB)
  _Float16* qb   = (_Float16*)(ws + (size_t)(4u << 20));     // 64*1024*32 (4MB)
  _Float16* kbuf = (_Float16*)(ws + (size_t)(8u << 20));     // 64*1024*32 (4MB)
  _Float16* vT   = (_Float16*)(ws + (size_t)(12u << 20));    // 64*32*1024 (4MB)
  _Float16* ob   = (_Float16*)(ws + (size_t)(16u << 20));    // 8192*256   (4MB)
  _Float16* wqkv = (_Float16*)(ws + (size_t)(20u << 20));    // 768*256
  _Float16* wout = (_Float16*)(ws + (size_t)(20u << 20) + 768 * 256 * 2);

  {
    const int na = 768 * 256, nb = 256 * 256;
    const int n = na + nb;
    cvt_kernel<<<(n + 255) / 256, 256, 0, stream>>>(qkv_w, na, out_w, nb,
                                                    wqkv, wout);
  }
  poscnn_kernel<<<(8 * 1024 * 256) / 256, 256, 0, stream>>>(x, conv_w, conv_b, xp);
  qkv_gemm_kernel<<<(512 * 12) / 8, 256, 0, stream>>>(xp, wqkv, qb, kbuf, vT);
  attn_kernel<<<1024, 128, 0, stream>>>(qb, kbuf, vT, ob);
  out_gemm_kernel<<<(512 * 4) / 8, 256, 0, stream>>>(ob, wout, out_b, out);
}
